// Attractor_60782377173273
// MI455X (gfx1250) — matrix operand order repaired and checked by reading, not compiled, not run on hardware
//
#include <hip/hip_runtime.h>
#include <math.h>

typedef __attribute__((ext_vector_type(16))) __bf16        v16bf;
typedef __attribute__((ext_vector_type(8)))  float         v8f;
typedef __attribute__((ext_vector_type(4)))  unsigned int  u32x4;

union FragBF { v16bf v; u32x4 q[2]; unsigned short s[16]; };

#define BL       32768
#define C_DIM    256
#define N_DIM    512
#define K_ITERS  15
#define M_TILE   32
#define THREADS  256
#define OUT_OFF  ((size_t)8 * 4096 * 256)

// ws layout (ushort elems): Ws_sym bf16 (512x512, row n, col k), then w_in_w bf16
// (512x256), then w_out_w bf16 (256x512)
#define WS_WSYM  0
#define WS_WIN   (512 * 512)
#define WS_WOUT  (512 * 512 + 512 * 256)

// LDS row strides (bf16 elems), padded so 16 lanes doing ds_load_b128 down a
// column hit distinct bank groups (stride*2 bytes -> lane bank step = 4)
#define SA_STRIDE 520   // 512 + 8
#define SX_STRIDE 264   // 256 + 8

__device__ __forceinline__ unsigned short f2bf(float f) {
  unsigned int u = __float_as_uint(f);
  u += 0x7fffu + ((u >> 16) & 1u);          // round-to-nearest-even
  return (unsigned short)(u >> 16);
}

__device__ __forceinline__ float fast_tanh(float x) {
#if __has_builtin(__builtin_amdgcn_tanhf)
  return __builtin_amdgcn_tanhf(x);         // V_TANH_F32 (TRANS unit)
#else
  return tanhf(x);
#endif
}

__global__ void Attractor_prep_weights(const float* __restrict__ W,
                                       const float* __restrict__ w_in_w,
                                       const float* __restrict__ w_out_w,
                                       unsigned short* __restrict__ ws) {
  const int i = blockIdx.x * blockDim.x + threadIdx.x;   // 0 .. 262143
  if (i < 512 * 512) {
    const int n = i >> 9, k = i & 511;
    ws[WS_WSYM + i] = f2bf(0.5f * (W[i] + W[k * 512 + n]));   // Ws = (W+W^T)/2
  }
  if (i < 512 * 256) ws[WS_WIN + i]  = f2bf(w_in_w[i]);
  if (i < 256 * 512) ws[WS_WOUT + i] = f2bf(w_out_w[i]);
}

__global__ __launch_bounds__(THREADS, 1)
void Attractor_main(const float* __restrict__ x,
                    const float* __restrict__ w_in_b,
                    const float* __restrict__ bvec,
                    const float* __restrict__ w_out_b,
                    const unsigned short* __restrict__ ws,
                    float* __restrict__ out) {
  __shared__ unsigned short sA[M_TILE * SA_STRIDE];   // bf16 state a (32 x 512)
  __shared__ unsigned short sX[M_TILE * SX_STRIDE];   // bf16 x tile  (32 x 256)

  const unsigned short* wsym = ws + WS_WSYM;
  const unsigned short* win  = ws + WS_WIN;
  const unsigned short* wout = ws + WS_WOUT;

  const int tid  = threadIdx.x;
  const int lane = tid & 31;
  const int wave = tid >> 5;          // 0..7
  const int wm   = wave & 1;          // 16-row half of the tile
  const int wn   = wave >> 1;         // 0..3 : 128-col group of N=512
  const int l15  = lane & 15;
  const int lh   = lane >> 4;         // 0/1
  const int arow = wm * 16 + l15;     // A-fragment row (lane -> M)
  const int rbas = wm * 16 + lh * 8;  // C-fragment base row (VGPR i adds 0..7)

  const size_t row0 = (size_t)blockIdx.x * M_TILE;

  // hint Ws into cache for the iteration loop
  __builtin_prefetch(wsym + (size_t)(wn * 128 + l15) * N_DIM, 0, 0);

  // ---- stage x tile as bf16 into LDS ----
  const float* xg = x + row0 * C_DIM;
  for (int i = tid; i < M_TILE * C_DIM; i += THREADS)
    sX[(i >> 8) * SX_STRIDE + (i & 255)] = f2bf(xg[i]);
  __syncthreads();

  // ---- GEMM1: cb = x @ w_in_w^T  (wave: 16 rows x 128 cols of N=512) ----
  v8f cb[8] = {};
  for (int kc = 0; kc < C_DIM / 32; ++kc) {
    FragBF A;
    const int ak = kc * 32 + lh * 8;
    A.q[0] = *(const u32x4*)&sX[arow * SX_STRIDE + ak];
    A.q[1] = *(const u32x4*)&sX[arow * SX_STRIDE + ak + 16];
    const int kb = kc * 32 + lh * 8;
#pragma unroll
    for (int nt = 0; nt < 8; ++nt) {
      const int col = wn * 128 + nt * 16 + l15;
      FragBF Bf;
      const u32x4* wp = (const u32x4*)(win + (size_t)col * C_DIM + kb);
      Bf.q[0] = wp[0]; Bf.q[1] = wp[2];
      cb[nt] = __builtin_amdgcn_wmma_f32_16x16x32_bf16(
          false, A.v, false, Bf.v, (short)0, cb[nt], false, false);
    }
  }

  // cb += w_in_b + b ; a1 = tanh(cb) -> LDS (a0 == 0)
#pragma unroll
  for (int nt = 0; nt < 8; ++nt) {
    const int col = wn * 128 + nt * 16 + l15;
    cb[nt] += (w_in_b[col] + bvec[col]);
#pragma unroll
    for (int i = 0; i < 8; ++i)
      sA[(rbas + i) * SA_STRIDE + col] = f2bf(fast_tanh(cb[nt][i]));
  }

  // ---- 14 remaining iterations: a = tanh(a @ Ws^T + cb) ----
  for (int it = 1; it < K_ITERS; ++it) {
    __syncthreads();                       // new a visible (RAW)
    v8f acc[8];
#pragma unroll
    for (int nt = 0; nt < 8; ++nt) acc[nt] = cb[nt];

    for (int kc = 0; kc < N_DIM / 32; ++kc) {
      FragBF A;
      const int ak = kc * 32 + lh * 8;
      A.q[0] = *(const u32x4*)&sA[arow * SA_STRIDE + ak];
      A.q[1] = *(const u32x4*)&sA[arow * SA_STRIDE + ak + 16];
      const int kb = kc * 32 + lh * 8;
#pragma unroll
      for (int nt = 0; nt < 8; ++nt) {
        const int col = wn * 128 + nt * 16 + l15;
        FragBF Bf;
        const u32x4* wp = (const u32x4*)(wsym + (size_t)col * N_DIM + kb);
        Bf.q[0] = wp[0]; Bf.q[1] = wp[2];
        acc[nt] = __builtin_amdgcn_wmma_f32_16x16x32_bf16(
            false, A.v, false, Bf.v, (short)0, acc[nt], false, false);
      }
    }
    __syncthreads();                       // all reads of old a done (WAR)
#pragma unroll
    for (int nt = 0; nt < 8; ++nt) {
      const int col = wn * 128 + nt * 16 + l15;
#pragma unroll
      for (int i = 0; i < 8; ++i)
        sA[(rbas + i) * SA_STRIDE + col] = f2bf(fast_tanh(acc[nt][i]));
    }
  }
  __syncthreads();                         // final a visible

  // ---- GEMM3: y = a @ w_out_w^T (32 x 256); wave: 16 rows x 64 cols ----
  v8f yac[4] = {};
  for (int kc = 0; kc < N_DIM / 32; ++kc) {
    FragBF A;
    const int ak = kc * 32 + lh * 8;
    A.q[0] = *(const u32x4*)&sA[arow * SA_STRIDE + ak];
    A.q[1] = *(const u32x4*)&sA[arow * SA_STRIDE + ak + 16];
    const int kb = kc * 32 + lh * 8;
#pragma unroll
    for (int nt = 0; nt < 4; ++nt) {
      const int col = wn * 64 + nt * 16 + l15;
      FragBF Bf;
      const u32x4* wp = (const u32x4*)(wout + (size_t)col * N_DIM + kb);
      Bf.q[0] = wp[0]; Bf.q[1] = wp[2];
      yac[nt] = __builtin_amdgcn_wmma_f32_16x16x32_bf16(
          false, A.v, false, Bf.v, (short)0, yac[nt], false, false);
    }
  }

  // ---- fused epilogue: out0 = y, out1 = x - y ----
#pragma unroll
  for (int nt = 0; nt < 4; ++nt) {
    const int col = wn * 64 + nt * 16 + l15;
    const float ob = w_out_b[col];
#pragma unroll
    for (int i = 0; i < 8; ++i) {
      const size_t oi = (row0 + rbas + i) * C_DIM + col;
      const float y  = yac[nt][i] + ob;
      out[oi]           = y;
      out[OUT_OFF + oi] = x[oi] - y;
    }
  }
}

extern "C" void kernel_launch(void* const* d_in, const int* in_sizes, int n_in,
                              void* d_out, int out_size, void* d_ws, size_t ws_size,
                              hipStream_t stream) {
  (void)in_sizes; (void)n_in; (void)out_size; (void)ws_size;
  const float* x       = (const float*)d_in[0];
  const float* w_in_w  = (const float*)d_in[1];
  const float* w_in_b  = (const float*)d_in[2];
  const float* W       = (const float*)d_in[3];
  const float* b       = (const float*)d_in[4];
  const float* w_out_w = (const float*)d_in[5];
  const float* w_out_b = (const float*)d_in[6];
  unsigned short* ws   = (unsigned short*)d_ws;
  float* out           = (float*)d_out;

  Attractor_prep_weights<<<1024, 256, 0, stream>>>(W, w_in_w, w_out_w, ws);
  Attractor_main<<<BL / M_TILE, THREADS, 0, stream>>>(x, w_in_b, b, w_out_b, ws, out);
}
